// SpatialPerceptionBlock_41437844472101
// MI455X (gfx1250) — matrix operand order repaired and checked by reading, not compiled, not run on hardware
//
#include <hip/hip_runtime.h>
#include <hip/hip_bf16.h>
#include <math.h>

// ---------------------------------------------------------------------------
// SpatialPerceptionBlock for MI455X (gfx1250, wave32, WMMA f16->f32, TDM)
//   B=2, CIN=64, COUT=128, H=W=256, K=9 taps, RED=8
// ---------------------------------------------------------------------------

typedef _Float16 half4v  __attribute__((ext_vector_type(4)));
typedef _Float16 half8v  __attribute__((ext_vector_type(8)));
typedef _Float16 half16v __attribute__((ext_vector_type(16)));
typedef float    float8v __attribute__((ext_vector_type(8)));
typedef unsigned int u32x4_t __attribute__((ext_vector_type(4)));
typedef int          i32x4_t __attribute__((ext_vector_type(4)));
typedef int          i32x8_t __attribute__((ext_vector_type(8)));

#define HW   65536
#define WID  256
#define CIN  64
#define COUT 128
#define NTAP 9
#define EPS  1e-5f
#define LDP  72   // padded LDS row (halves): 144B stride -> conflict-free b128

// TDM available? (device pass only; arity differs between toolchains)
#if defined(__HIP_DEVICE_COMPILE__) && __has_builtin(__builtin_amdgcn_tensor_load_to_lds)
#define USE_TDM 1
#else
#define USE_TDM 0
#endif

// ---------------------------------------------------------------------------
// Kernel 0: pre-convert w_def (128,64,3,3) f32 -> tap-major f16 [t][n][c]
// (each tap slice is a contiguous 16KB tile => TDM-friendly)
// ---------------------------------------------------------------------------
__global__ void __launch_bounds__(256)
k_prep_wdef(const float* __restrict__ w_def, _Float16* __restrict__ wdefs) {
    const int i = blockIdx.x * 256 + threadIdx.x;      // 9*128*64 = 73728
    if (i < NTAP * COUT * CIN) {
        const int t = i / (COUT * CIN);
        const int r = i % (COUT * CIN);
        const int n = r >> 6, c = r & 63;
        wdefs[i] = (_Float16)w_def[(size_t)n * (CIN * 9) + c * 9 + t];
    }
}

// ---------------------------------------------------------------------------
// Kernel 1: 3x3 offset conv, 64 -> 18 channels, pad=1, fp32 VALU.
// ---------------------------------------------------------------------------
__global__ void __launch_bounds__(256)
k_offset_conv(const float* __restrict__ x, const float* __restrict__ w_off,
              const float* __restrict__ b_off, float* __restrict__ offs) {
    __shared__ float Wl[18 * 64 * 9];   // 41472 B
    const int tid = threadIdx.x;
    for (int i = tid; i < 18 * 64 * 9; i += 256) Wl[i] = w_off[i];
    __syncthreads();

    const int p  = blockIdx.x * 256 + tid;        // 0..131071
    const int b  = p >> 16;
    const int y  = (p >> 8) & 255;
    const int xx = p & 255;
    const float* xb = x + (size_t)b * CIN * HW;

    float acc[18];
#pragma unroll
    for (int o = 0; o < 18; ++o) acc[o] = b_off[o];

    for (int c = 0; c < CIN; ++c) {
        const float* xc = xb + (size_t)c * HW;
        float v[9];
#pragma unroll
        for (int j = 0; j < 9; ++j) {
            const int yy = y + j / 3 - 1;
            const int xw = xx + j % 3 - 1;
            v[j] = (yy >= 0 && yy < WID && xw >= 0 && xw < WID) ? xc[yy * WID + xw] : 0.f;
        }
        const float* wc = Wl + c * 9;
#pragma unroll
        for (int o = 0; o < 18; ++o) {
            const float* w9 = wc + o * (64 * 9);
#pragma unroll
            for (int j = 0; j < 9; ++j) acc[o] = fmaf(v[j], w9[j], acc[o]);
        }
    }
    float* ob = offs + (size_t)b * 18 * HW + y * WID + xx;
#pragma unroll
    for (int o = 0; o < 18; ++o) ob[(size_t)o * HW] = acc[o];   // coalesced per channel
}

// ---------------------------------------------------------------------------
// Kernel 2: deformable conv GEMM via WMMA f16 (M=128px/WG, N=128, K=576).
// Per tap: TDM DMAs the 16KB weight slice into padded LDS rows (overlapped
// with bilinear-gather A staging), then 8 waves x 16 WMMAs.
// ---------------------------------------------------------------------------
__global__ void __launch_bounds__(256)
k_deform(const float* __restrict__ x, const float* __restrict__ offs,
         const _Float16* __restrict__ wdefs,   // [t][n][c] f16, tap-major
         const float* __restrict__ b_def,
         const float* __restrict__ bng, const float* __restrict__ bnb,
         const float* __restrict__ bnm, const float* __restrict__ bnv,
         float* __restrict__ out, float* __restrict__ pooled) {
    __shared__ __align__(16) _Float16 Alds[128 * LDP];   // 18 KB
    __shared__ __align__(16) _Float16 Blds[128 * LDP];   // 18 KB
    __shared__ int   midx[128][4];
    __shared__ float mwt[128][4];
    __shared__ float sA[128], sB[128], psum[128];

    const int tid = threadIdx.x;
    const int p0  = blockIdx.x * 128;          // 128 consecutive pixels (same row)
    const int b   = p0 >> 16;
    const int y   = (p0 >> 8) & 255;
    const int xb0 = p0 & 255;

    if (tid < 128) {
        const float s = bng[tid] * rsqrtf(bnv[tid] + EPS);
        sA[tid]   = s;
        sB[tid]   = b_def[tid] * s + bnb[tid] - bnm[tid] * s;  // bias folded in shift
        psum[tid] = 0.f;
    }

    float8v acc[8];
#pragma unroll
    for (int i = 0; i < 8; ++i)
#pragma unroll
        for (int j = 0; j < 8; ++j) acc[i][j] = 0.f;

    const float* xplane = x + (size_t)b * CIN * HW;
    const int wave = tid >> 5, lane = tid & 31, g = lane >> 4;
    const int mrow = wave * 16 + (lane & 15);

#if USE_TDM
    // low-32 bits of the flat shared-pointer == LDS byte offset
    const unsigned lds_base = (unsigned)(size_t)(&Blds[0]);
#endif

    for (int t = 0; t < NTAP; ++t) {
        __syncthreads();   // previous tap's WMMA reads of Alds/Blds are done

#if USE_TDM
        // ---- TDM: DMA tap t's weight slice (128x64 f16, contiguous 16KB)
        // into Blds with hardware padding: +4 DWORDs after every 32 DWORDs
        // => 144B row stride == LDP halves, matching the fragment reads.
        if (wave == 0) {
            const unsigned long long ga =
                (unsigned long long)(size_t)wdefs + (unsigned long long)t * 16384ull;
            u32x4_t g0;
            g0[0] = 1u;                                    // count=1 (valid user D#)
            g0[1] = lds_base;                              // lds_addr (bytes)
            g0[2] = (unsigned)(ga & 0xffffffffu);          // global_addr[31:0]
            g0[3] = (unsigned)((ga >> 32) & 0x01ffffffu)   // global_addr[56:32]
                  | 0x80000000u;                           // type=2 ("image")
            i32x8_t g1;
            g1[0] = (1 << 16)        // data_size = 1 -> 2-byte elements
                  | (1 << 20)        // pad_enable
                  | (4 << 22)        // pad_interval: 32 DWORDs
                  | (3 << 25);       // pad_amount:   4 DWORDs
            g1[1] = (int)(8192u << 16);   // tensor_dim0 = 8192 (lo16 in [31:16])
            g1[2] = (1 << 16);            // tensor_dim0 hi=0 | tensor_dim1 = 1
            g1[3] = (int)(8192u << 16);   // tensor_dim1 hi=0 | tile_dim0 = 8192
            g1[4] = 1;                    // tile_dim1 = 1 | tile_dim2 = 0
            g1[5] = 8192;                 // tensor_dim0_stride lo32
            g1[6] = 0;                    // stride hi | tensor_dim1_stride lo
            g1[7] = 0;
            const i32x4_t gz = {0, 0, 0, 0};
#if defined(__clang_major__) && __clang_major__ >= 23
            const i32x8_t gz8 = {0, 0, 0, 0, 0, 0, 0, 0};
            __builtin_amdgcn_tensor_load_to_lds(g0, g1, gz, gz, gz8, 0);
#else
            __builtin_amdgcn_tensor_load_to_lds(g0, g1, gz, gz, 0);
#endif
        }
#else
        // ---- fallback: manual weight staging
        for (int i = 0; i < 32; ++i) {
            const int e = i * 256 + tid;
            const int n = e >> 6, c = e & 63;
            Blds[n * LDP + c] = wdefs[t * (COUT * CIN) + n * CIN + c];
        }
#endif

        // ---- per-pixel bilinear metadata for tap t
        if (tid < 128) {
            const int p  = tid;
            const int xx = xb0 + p;
            const float dy = offs[((size_t)(b * 18 + 2 * t)) * HW + y * WID + xx];
            const float dx = offs[((size_t)(b * 18 + 2 * t + 1)) * HW + y * WID + xx];
            const float py = (float)(y + t / 3 - 1) + dy;
            const float px = (float)(xx + t % 3 - 1) + dx;
            const float y0f = floorf(py), x0f = floorf(px);
            const int y0 = (int)y0f, x0 = (int)x0f;
            const float wy1 = py - y0f, wx1 = px - x0f;
            const float wy0 = 1.f - wy1, wx0 = 1.f - wx1;
            const bool vy0 = (y0 >= 0) & (y0 < WID), vy1 = (y0 + 1 >= 0) & (y0 + 1 < WID);
            const bool vx0 = (x0 >= 0) & (x0 < WID), vx1 = (x0 + 1 >= 0) & (x0 + 1 < WID);
            const int cy0 = min(max(y0, 0), WID - 1), cy1 = min(max(y0 + 1, 0), WID - 1);
            const int cx0 = min(max(x0, 0), WID - 1), cx1 = min(max(x0 + 1, 0), WID - 1);
            midx[p][0] = cy0 * WID + cx0;  midx[p][1] = cy0 * WID + cx1;
            midx[p][2] = cy1 * WID + cx0;  midx[p][3] = cy1 * WID + cx1;
            mwt[p][0] = (vy0 && vx0) ? wy0 * wx0 : 0.f;
            mwt[p][1] = (vy0 && vx1) ? wy0 * wx1 : 0.f;
            mwt[p][2] = (vy1 && vx0) ? wy1 * wx0 : 0.f;
            mwt[p][3] = (vy1 && vx1) ? wy1 * wx1 : 0.f;
        }
        __syncthreads();

        // ---- stage sampled A tile: thread owns pixel p, contiguous 32-ch range,
        //      4 halves per LDS store (b64)
        {
            const int p  = tid & 127, hi = tid >> 7;
            const int i0 = midx[p][0], i1 = midx[p][1], i2 = midx[p][2], i3 = midx[p][3];
            const float w0 = mwt[p][0], w1 = mwt[p][1], w2 = mwt[p][2], w3 = mwt[p][3];
            for (int i = 0; i < 32; i += 4) {
                const int c0 = hi * 32 + i;
                half4v v4;
#pragma unroll
                for (int j = 0; j < 4; ++j) {
                    const float* xc = xplane + (size_t)(c0 + j) * HW;
                    v4[j] = (_Float16)(w0 * xc[i0] + w1 * xc[i1] +
                                       w2 * xc[i2] + w3 * xc[i3]);
                }
                *(half4v*)&Alds[p * LDP + c0] = v4;
            }
        }

#if USE_TDM
        if (wave == 0) __builtin_amdgcn_s_wait_tensorcnt(0);
#endif
        __syncthreads();

        // ---- WMMA: A frag = 16-K run per lane; B frag = 16-K run per lane
        half16v a0, a1;
        ((half8v*)&a0)[0] = *(const half8v*)&Alds[mrow * LDP + g * 16];
        ((half8v*)&a0)[1] = *(const half8v*)&Alds[mrow * LDP + g * 16 + 8];
        ((half8v*)&a1)[0] = *(const half8v*)&Alds[mrow * LDP + 32 + g * 16];
        ((half8v*)&a1)[1] = *(const half8v*)&Alds[mrow * LDP + 32 + g * 16 + 8];
#pragma unroll
        for (int nt = 0; nt < 8; ++nt) {
            const int n = nt * 16 + (lane & 15);
            half16v b0, b1;
            ((half8v*)&b0)[0] = *(const half8v*)&Blds[n * LDP + g * 16];
            ((half8v*)&b0)[1] = *(const half8v*)&Blds[n * LDP + g * 16 + 8];
            ((half8v*)&b1)[0] = *(const half8v*)&Blds[n * LDP + 32 + g * 16];
            ((half8v*)&b1)[1] = *(const half8v*)&Blds[n * LDP + 32 + g * 16 + 8];
            acc[nt] = __builtin_amdgcn_wmma_f32_16x16x32_f16(false, a0, false, b0,
                                                             (short)0, acc[nt], false, false);
            acc[nt] = __builtin_amdgcn_wmma_f32_16x16x32_f16(false, a1, false, b1,
                                                             (short)0, acc[nt], false, false);
        }
    }

    // ---- epilogue: BN, contiguous 8-pixel stores per lane, pooled reduction
    const int mbase = wave * 16 + 8 * g;   // accumulator r -> pixel mbase+r
#pragma unroll
    for (int nt = 0; nt < 8; ++nt) {
        const int n = nt * 16 + (lane & 15);
        const float s = sA[n], sh = sB[n];
        float r[8];
#pragma unroll
        for (int i = 0; i < 8; ++i) r[i] = acc[nt][i] * s + sh;
        float* op = out + ((size_t)(b * COUT + n)) * HW + y * WID + xb0 + mbase;
        *(float4*)op       = make_float4(r[0], r[1], r[2], r[3]);
        *(float4*)(op + 4) = make_float4(r[4], r[5], r[6], r[7]);
        const float lsum = ((r[0] + r[1]) + (r[2] + r[3])) + ((r[4] + r[5]) + (r[6] + r[7]));
        atomicAdd(&psum[n], lsum);   // ds_add_f32
    }
    __syncthreads();
    if (tid < 128) atomicAdd(&pooled[b * COUT + tid], psum[tid]);  // global atomic
}

// ---------------------------------------------------------------------------
// Kernel 3: SE gate (tiny). pooled mean -> 1x1 -> BN -> GELU(exact) ->
// @softmax(phi) -> sigmoid -> dw[2][128]
// ---------------------------------------------------------------------------
__global__ void k_se(const float* __restrict__ pooled,
                     const float* __restrict__ se_w, const float* __restrict__ se_b,
                     const float* __restrict__ sg, const float* __restrict__ sb,
                     const float* __restrict__ sm, const float* __restrict__ sv,
                     const float* __restrict__ phi, float* __restrict__ dw) {
    __shared__ float pl[128], tl[16], rmax[16], rsum[16];
    const int tid = threadIdx.x;
    for (int b = 0; b < 2; ++b) {
        if (tid < 128) pl[tid] = pooled[b * COUT + tid] * (1.f / (float)HW);
        __syncthreads();
        if (tid < 16) {
            float a = se_b[tid];
            for (int c = 0; c < 128; ++c) a += pl[c] * se_w[tid * 128 + c];
            const float s = sg[tid] * rsqrtf(sv[tid] + EPS);
            float tv = a * s + sb[tid] - sm[tid] * s;
            tv = 0.5f * tv * (1.f + erff(tv * 0.70710678118654752f));  // exact GELU
            tl[tid] = tv;
            float mx = -1e30f;
            for (int c = 0; c < 128; ++c) mx = fmaxf(mx, phi[tid * 128 + c]);
            float ss = 0.f;
            for (int c = 0; c < 128; ++c) ss += expf(phi[tid * 128 + c] - mx);
            rmax[tid] = mx; rsum[tid] = ss;
        }
        __syncthreads();
        if (tid < 128) {
            float a = 0.f;
            for (int r = 0; r < 16; ++r)
                a += tl[r] * expf(phi[r * 128 + tid] - rmax[r]) / rsum[r];
            dw[b * COUT + tid] = 1.f / (1.f + expf(-a));
        }
        __syncthreads();
    }
}

// ---------------------------------------------------------------------------
// Kernel 4: 1x1 shortcut GEMM via WMMA + gate + BN + ReLU, in place on d_out
// ---------------------------------------------------------------------------
__global__ void __launch_bounds__(256)
k_final(const float* __restrict__ x, const float* __restrict__ ds_w,
        const float* __restrict__ dg, const float* __restrict__ db,
        const float* __restrict__ dm, const float* __restrict__ dv,
        const float* __restrict__ dw, float* __restrict__ out) {
    __shared__ __align__(16) _Float16 Xl[128 * LDP];
    __shared__ __align__(16) _Float16 Bl[128 * LDP];
    __shared__ float sA[128], sB[128], gate[128];

    const int tid = threadIdx.x;
    const int p0  = blockIdx.x * 128;
    const int b   = p0 >> 16;
    const int y   = (p0 >> 8) & 255;
    const int xb0 = p0 & 255;

    if (tid < 128) {
        const float s = dg[tid] * rsqrtf(dv[tid] + EPS);
        sA[tid] = s; sB[tid] = db[tid] - dm[tid] * s;
        gate[tid] = dw[b * COUT + tid];
    }
    for (int i = 0; i < 32; ++i) {                       // B: ds_w[n][c]
        const int e = i * 256 + tid;
        const int n = e >> 6, c = e & 63;
        Bl[n * LDP + c] = (_Float16)ds_w[n * CIN + c];
    }
    {                                                    // A: center pixels
        const int p = tid & 127, hi = tid >> 7;
        const int pix = y * WID + xb0 + p;
        for (int i = 0; i < 32; i += 4) {
            const int c0 = hi * 32 + i;
            half4v v4;
#pragma unroll
            for (int j = 0; j < 4; ++j)
                v4[j] = (_Float16)x[((size_t)(b * CIN + c0 + j)) * HW + pix];
            *(half4v*)&Xl[p * LDP + c0] = v4;
        }
    }
    __syncthreads();

    const int wave = tid >> 5, lane = tid & 31, g = lane >> 4;
    const int mrow = wave * 16 + (lane & 15);
    half16v a0, a1;
    ((half8v*)&a0)[0] = *(const half8v*)&Xl[mrow * LDP + g * 16];
    ((half8v*)&a0)[1] = *(const half8v*)&Xl[mrow * LDP + g * 16 + 8];
    ((half8v*)&a1)[0] = *(const half8v*)&Xl[mrow * LDP + 32 + g * 16];
    ((half8v*)&a1)[1] = *(const half8v*)&Xl[mrow * LDP + 32 + g * 16 + 8];

    const int mbase = wave * 16 + 8 * g;
#pragma unroll
    for (int nt = 0; nt < 8; ++nt) {
        const int n = nt * 16 + (lane & 15);
        half16v b0, b1;
        ((half8v*)&b0)[0] = *(const half8v*)&Bl[n * LDP + g * 16];
        ((half8v*)&b0)[1] = *(const half8v*)&Bl[n * LDP + g * 16 + 8];
        ((half8v*)&b1)[0] = *(const half8v*)&Bl[n * LDP + 32 + g * 16];
        ((half8v*)&b1)[1] = *(const half8v*)&Bl[n * LDP + 32 + g * 16 + 8];
        float8v acc;
#pragma unroll
        for (int j = 0; j < 8; ++j) acc[j] = 0.f;
        acc = __builtin_amdgcn_wmma_f32_16x16x32_f16(false, a0, false, b0,
                                                     (short)0, acc, false, false);
        acc = __builtin_amdgcn_wmma_f32_16x16x32_f16(false, a1, false, b1,
                                                     (short)0, acc, false, false);

        const float s = sA[n], sh = sB[n], gn = gate[n];
        float* op = out + ((size_t)(b * COUT + n)) * HW + y * WID + xb0 + mbase;
        const float4 o0 = *(const float4*)op;
        const float4 o1 = *(const float4*)(op + 4);
        float4 r0, r1;
        r0.x = fmaxf(o0.x * gn + (acc[0] * s + sh), 0.f);
        r0.y = fmaxf(o0.y * gn + (acc[1] * s + sh), 0.f);
        r0.z = fmaxf(o0.z * gn + (acc[2] * s + sh), 0.f);
        r0.w = fmaxf(o0.w * gn + (acc[3] * s + sh), 0.f);
        r1.x = fmaxf(o1.x * gn + (acc[4] * s + sh), 0.f);
        r1.y = fmaxf(o1.y * gn + (acc[5] * s + sh), 0.f);
        r1.z = fmaxf(o1.z * gn + (acc[6] * s + sh), 0.f);
        r1.w = fmaxf(o1.w * gn + (acc[7] * s + sh), 0.f);
        *(float4*)op       = r0;
        *(float4*)(op + 4) = r1;
    }
}

// ---------------------------------------------------------------------------
extern "C" void kernel_launch(void* const* d_in, const int* in_sizes, int n_in,
                              void* d_out, int out_size, void* d_ws, size_t ws_size,
                              hipStream_t stream) {
    const float* x       = (const float*)d_in[0];
    const float* w_off   = (const float*)d_in[1];
    const float* b_off   = (const float*)d_in[2];
    const float* w_def   = (const float*)d_in[3];
    const float* b_def   = (const float*)d_in[4];
    const float* bng     = (const float*)d_in[5];
    const float* bnb     = (const float*)d_in[6];
    const float* bnm     = (const float*)d_in[7];
    const float* bnv     = (const float*)d_in[8];
    const float* se_w    = (const float*)d_in[9];
    const float* se_b    = (const float*)d_in[10];
    const float* se_bng  = (const float*)d_in[11];
    const float* se_bnb  = (const float*)d_in[12];
    const float* se_bnm  = (const float*)d_in[13];
    const float* se_bnv  = (const float*)d_in[14];
    const float* phi     = (const float*)d_in[15];
    const float* ds_w    = (const float*)d_in[16];
    const float* ds_bng  = (const float*)d_in[17];
    const float* ds_bnb  = (const float*)d_in[18];
    const float* ds_bnm  = (const float*)d_in[19];
    const float* ds_bnv  = (const float*)d_in[20];
    float* outp = (float*)d_out;

    // ws layout: offsets (2*18*HW f32) | pooled (256 f32) | dw (256 f32) |
    //            wdefs f16 tap-major (9*128*64)
    float*     offs   = (float*)d_ws;
    float*     pooled = (float*)((char*)d_ws + (size_t)2 * 18 * HW * sizeof(float));
    float*     dwv    = pooled + 256;
    _Float16*  wdefs  = (_Float16*)(dwv + 256);

    (void)hipMemsetAsync(pooled, 0, 256 * sizeof(float), stream);

    k_prep_wdef<<<288, 256, 0, stream>>>(w_def, wdefs);
    k_offset_conv<<<512, 256, 0, stream>>>(x, w_off, b_off, offs);
    k_deform<<<1024, 256, 0, stream>>>(x, offs, wdefs, b_def,
                                       bng, bnb, bnm, bnv, outp, pooled);
    k_se<<<1, 128, 0, stream>>>(pooled, se_w, se_b,
                                se_bng, se_bnb, se_bnm, se_bnv, phi, dwv);
    k_final<<<1024, 256, 0, stream>>>(x, ds_w, ds_bng, ds_bnb, ds_bnm, ds_bnv,
                                      dwv, outp);
}
